// AdvantageNetwork_61976378081551
// MI455X (gfx1250) — matrix operand order repaired and checked by reading, not compiled, not run on hardware
//
#include <hip/hip_runtime.h>

typedef __attribute__((ext_vector_type(16))) __bf16 v16bf;
typedef __attribute__((ext_vector_type(8)))  __bf16 v8bf;
typedef __attribute__((ext_vector_type(4)))  __bf16 v4bf;
typedef __attribute__((ext_vector_type(8)))  float  v8f;
typedef __attribute__((ext_vector_type(4)))  float  v4f;

#define LDX 264   // row stride (elems) for 256-wide bf16 LDS arrays (528B, 16B aligned)
#define LDW 40    // row stride for 32-wide softmax-weight array
#define LDH 72    // row stride for 64-wide hidden array

typedef __attribute__((address_space(3))) float lds_float;

// ---- WMMA helper: D = A(16x32 bf16) x B(32x16 bf16) + C(f32) ----
__device__ __forceinline__ v8f wmma_bf16(v16bf a, v16bf b, v8f c) {
  return __builtin_amdgcn_wmma_f32_16x16x32_bf16(false, a, false, b, (short)0, c,
                                                 false, false);
}

// A-matrix fragment (16x32 bf16 tile, row-major source, row = m0+lane%16).
// Layout: lane L holds K = k0 + 16*(L>>4) + [0..15] contiguous (matches B).
// Two 16B loads per lane.
__device__ __forceinline__ v16bf load_A(const __bf16* base, int ld, int m0, int k0,
                                        int lane) {
  const __bf16* row = base + (m0 + (lane & 15)) * ld;
  int ks = k0 + ((lane >> 4) << 4);
  v8bf lo = *(const v8bf*)(row + ks);
  v8bf hi = *(const v8bf*)(row + ks + 8);
  v16bf r;
#pragma unroll
  for (int i = 0; i < 8; ++i) { r[i] = lo[i]; r[i + 8] = hi[i]; }
  return r;
}

// B-matrix fragment (32x16 bf16 tile) where B(k,n) = base[(n0+n)*ld + k0+k]
// (i.e. X * W^T with W row-major). Layout: lane L -> column n0+(L&15),
// K = k0 + 16*(L>>4) + [0..15] contiguous along the source row.
__device__ __forceinline__ v16bf load_Brow(const __bf16* base, int ld, int n0, int k0,
                                           int lane) {
  const __bf16* row = base + (n0 + (lane & 15)) * ld;
  int ks = k0 + ((lane >> 4) << 4);
  v8bf lo = *(const v8bf*)(row + ks);
  v8bf hi = *(const v8bf*)(row + ks + 8);
  v16bf r;
#pragma unroll
  for (int i = 0; i < 8; ++i) { r[i] = lo[i]; r[i + 8] = hi[i]; }
  return r;
}

// C fragment (v8f) -> bf16 LDS tile. lane L: col = n0+(L&15), rows m0+8*(L>>4)+r.
__device__ __forceinline__ void store_C_bf16(__bf16* base, int ld, int m0, int n0,
                                             int lane, v8f c) {
  int col = n0 + (lane & 15);
  int mrow = m0 + ((lane >> 4) << 3);
#pragma unroll
  for (int r = 0; r < 8; ++r) base[(mrow + r) * ld + col] = (__bf16)c[r];
}

// Async global->LDS copy of one 16-byte chunk (CDNA5 async path, ASYNCcnt).
__device__ __forceinline__ void async_copy_b128(uint32_t lds_byte_off,
                                                const float* gsrc) {
  asm volatile("global_load_async_to_lds_b128 %0, %1, off"
               :: "v"(lds_byte_off), "v"(gsrc) : "memory");
}

__device__ __forceinline__ void wait_asynccnt_8() {
  asm volatile("s_wait_asynccnt 0x8" ::: "memory");
}
__device__ __forceinline__ void wait_asynccnt_0() {
  asm volatile("s_wait_asynccnt 0x0" ::: "memory");
}

// ---- one-shot f32 -> bf16 weight conversion into workspace (L2-resident) ----
// layout: [Wk 65536][Wq 65536][Wv 65536][W1 16384][W2 1024]
__global__ void convert_weights_kernel(const float* __restrict__ Wk,
                                       const float* __restrict__ Wq,
                                       const float* __restrict__ Wv,
                                       const float* __restrict__ W1,
                                       const float* __restrict__ W2,
                                       __bf16* __restrict__ out) {
  int i = blockIdx.x * 256 + threadIdx.x;
  if (i < 65536) {
    out[i]          = (__bf16)Wk[i];
    out[65536 + i]  = (__bf16)Wq[i];
    out[131072 + i] = (__bf16)Wv[i];
  }
  if (i < 16384) out[196608 + i] = (__bf16)W1[i];
  if (i < 1024)  out[212992 + i] = (__bf16)W2[i];
}

__global__ __launch_bounds__(256) void adv_attn_kernel(
    const float* __restrict__ states, const float* __restrict__ actions,
    const __bf16* __restrict__ wts, float* __restrict__ advOut,
    float* __restrict__ wOut) {
  __shared__ float  stage[2][32 * 256]; // raw f32 x, filled by async DMA
  __shared__ __bf16 xs[32 * LDX];       // bf16 x, later reused for "weighted"
  __shared__ __bf16 Ks[32 * LDX];
  __shared__ __bf16 Qs[32 * LDX];
  __shared__ __bf16 Vs[32 * LDX];
  __shared__ float  Sb[32 * 33];        // attention scores f32
  __shared__ __bf16 wns[32 * LDW];      // softmax(w)/32 in bf16
  __shared__ __bf16 hs[32 * LDH];       // hidden layer bf16

  const int tid  = threadIdx.x;
  const int wv   = tid >> 5;            // wave id 0..7 (wave32)
  const int lane = tid & 31;
  const int b0   = blockIdx.x * 2;      // two batches per block, pipelined

  // ---- issue async HBM->LDS copies for BOTH batches up front.
  // x = [240 states | 16 actions] per row; 2048 16B-chunks per batch
  // (1920 states + 128 actions), 8 chunks per thread, coalesced.
#pragma unroll
  for (int it = 0; it < 2; ++it) {
    const int b = b0 + it;
    const float* sB = states  + (size_t)b * 32 * 240;
    const float* aB = actions + (size_t)b * 32 * 16;
    uint32_t stg = (uint32_t)(uintptr_t)(lds_float*)&stage[it][0];
#pragma unroll
    for (int j = 0; j < 8; ++j) {
      int c = tid + j * 256;
      const float* src;
      int ldsElem;
      if (c < 1920) {                 // states chunk
        int fs = c * 4;
        int row = fs / 240, col = fs % 240;
        src = sB + fs;
        ldsElem = row * 256 + col;
      } else {                        // actions chunk
        int fa = (c - 1920) * 4;
        int row = fa >> 4, col = 240 + (fa & 15);
        src = aB + fa;
        ldsElem = row * 256 + col;
      }
      async_copy_b128(stg + (uint32_t)ldsElem * 4u, src);
    }
  }

  for (int it = 0; it < 2; ++it) {
    const int b = b0 + it;
    // release this batch's staged data (in-order ASYNCcnt retire); batch 1's
    // HBM transfer keeps overlapping batch 0's whole compute.
    if (it == 0) wait_asynccnt_8(); else wait_asynccnt_0();
    __syncthreads();

    // ---- phase 1: stage f32 -> bf16 xs (vectorized LDS->LDS convert) ----
    for (int k = 0; k < 8; ++k) {
      int e4 = (tid + k * 256) * 4;
      int row = e4 >> 8, col = e4 & 255;
      v4f v = *(const v4f*)&stage[it][e4];
      v4bf o;
#pragma unroll
      for (int q = 0; q < 4; ++q) o[q] = (__bf16)v[q];
      *(v4bf*)&xs[row * LDX + col] = o;
    }
    __syncthreads();

    // ---- phase 2: K,Q,V = x * W^T. 48 (matrix, n-tile) pairs over 8 waves.
    for (int p = wv; p < 48; p += 8) {
      int mat = p >> 4, nt = p & 15;
      const __bf16* W = wts + mat * 65536;
      v8f c0 = {}; v8f c1 = {};
#pragma unroll
      for (int k = 0; k < 8; ++k) {
        v16bf bf = load_Brow(W, 256, nt * 16, k * 32, lane);
        v16bf a0 = load_A(xs, LDX, 0,  k * 32, lane);
        v16bf a1 = load_A(xs, LDX, 16, k * 32, lane);
        c0 = wmma_bf16(a0, bf, c0);
        c1 = wmma_bf16(a1, bf, c1);
      }
      __bf16* dst = (mat == 0) ? Ks : (mat == 1) ? Qs : Vs;
      store_C_bf16(dst, LDX, 0,  nt * 16, lane, c0);
      store_C_bf16(dst, LDX, 16, nt * 16, lane, c1);
    }
    __syncthreads();

    // ---- phase 3: S[i][j] = <K[i],Q[j]> / 16.  4 tiles on waves 0..3 ----
    if (wv < 4) {
      int mt = wv >> 1, nt = wv & 1;
      v8f c = {};
#pragma unroll
      for (int k = 0; k < 8; ++k) {
        v16bf a  = load_A(Ks, LDX, mt * 16, k * 32, lane);
        v16bf bf = load_Brow(Qs, LDX, nt * 16, k * 32, lane); // B(k,n)=Q[n][k]
        c = wmma_bf16(a, bf, c);
      }
      int col  = nt * 16 + (lane & 15);
      int mrow = mt * 16 + ((lane >> 4) << 3);
#pragma unroll
      for (int r = 0; r < 8; ++r) Sb[(mrow + r) * 33 + col] = c[r] * 0.0625f;
    }
    __syncthreads();

    // ---- phase 4: per-row softmax (32 lanes, one row each) ----
    if (tid < 32) {
      const float* srow = &Sb[tid * 33];
      float mx = srow[0];
      for (int j = 1; j < 32; ++j) mx = fmaxf(mx, srow[j]);
      float sum = 0.f;
      for (int j = 0; j < 32; ++j) sum += __expf(srow[j] - mx);
      float inv = 1.f / sum;
      float* wrow = wOut + ((size_t)b * 32 + tid) * 32;
      for (int j = 0; j < 32; ++j) {
        float w = __expf(srow[j] - mx) * inv;
        wrow[j] = w;                                  // output attention weights
        wns[tid * LDW + j] = (__bf16)(w * 0.03125f);  // w / N for next matmul
      }
    }
    __syncthreads();

    // ---- phase 5: weighted = wn * V (K=32 -> one WMMA per tile) ----
    for (int t = wv * 4; t < wv * 4 + 4; ++t) {
      int mt = t >> 4, nt = t & 15;
      v16bf a = load_A(wns, LDW, mt * 16, 0, lane);
      v16bf bf;                       // B(k=j,n=d) = V[j][d0+n]: column gather
      int col = nt * 16 + (lane & 15);
      int kb  = (lane >> 4) << 4;
#pragma unroll
      for (int e = 0; e < 16; ++e) bf[e] = Vs[(kb + e) * LDX + col];
      v8f c = {};
      c = wmma_bf16(a, bf, c);
      store_C_bf16(xs, LDX, mt * 16, nt * 16, lane, c);  // reuse xs buffer
    }
    __syncthreads();

    // ---- phase 6: h = leaky_relu(weighted * W1^T), 8 tiles on 8 waves ----
    {
      int mt = wv >> 2, nt = wv & 3;
      const __bf16* w1 = wts + 3 * 65536;
      v8f c = {};
#pragma unroll
      for (int k = 0; k < 8; ++k) {
        v16bf a  = load_A(xs, LDX, mt * 16, k * 32, lane);
        v16bf bf = load_Brow(w1, 256, nt * 16, k * 32, lane);
        c = wmma_bf16(a, bf, c);
      }
      int col  = nt * 16 + (lane & 15);
      int mrow = mt * 16 + ((lane >> 4) << 3);
#pragma unroll
      for (int r = 0; r < 8; ++r) {
        float v = c[r];
        v = v > 0.f ? v : 0.01f * v;
        hs[(mrow + r) * LDH + col] = (__bf16)v;
      }
    }
    __syncthreads();

    // ---- phase 7: adv = h * W2^T (2 tiles on waves 0,1) -> global ----
    if (wv < 2) {
      const __bf16* w2 = wts + 3 * 65536 + 16384;
      int mt = wv;
      v8f c = {};
#pragma unroll
      for (int k = 0; k < 2; ++k) {
        v16bf a  = load_A(hs, LDH, mt * 16, k * 32, lane);
        v16bf bf = load_Brow(w2, 64, 0, k * 32, lane);
        c = wmma_bf16(a, bf, c);
      }
      int col  = lane & 15;
      int mrow = mt * 16 + ((lane >> 4) << 3);
#pragma unroll
      for (int r = 0; r < 8; ++r)
        advOut[((size_t)b * 32 + mrow + r) * 16 + col] = c[r];
    }
    __syncthreads();   // protect xs/hs/Sb before next iteration reuses them
  }
}

extern "C" void kernel_launch(void* const* d_in, const int* in_sizes, int n_in,
                              void* d_out, int out_size, void* d_ws, size_t ws_size,
                              hipStream_t stream) {
  (void)in_sizes; (void)n_in; (void)out_size; (void)ws_size;
  const float* states  = (const float*)d_in[0];
  const float* actions = (const float*)d_in[1];
  const float* Wk = (const float*)d_in[2];
  const float* Wq = (const float*)d_in[3];
  const float* Wv = (const float*)d_in[4];
  const float* W1 = (const float*)d_in[5];
  const float* W2 = (const float*)d_in[6];

  __bf16* wbf = (__bf16*)d_ws;                 // 214016 bf16 = 428 KB scratch
  float* adv = (float*)d_out;                  // [4096,32,16]
  float* w   = adv + (size_t)4096 * 32 * 16;   // [4096,32,32,1]

  convert_weights_kernel<<<256, 256, 0, stream>>>(Wk, Wq, Wv, W1, W2, wbf);
  adv_attn_kernel<<<2048, 256, 0, stream>>>(states, actions, wbf, adv, w);
}
